// DotProductAttention_46737834115645
// MI455X (gfx1250) — compile-verified, matrix operand order checked by reading, not run on hardware
//
#include <hip/hip_runtime.h>

// ---- CDNA5 WMMA types (wave32) ----
typedef __attribute__((ext_vector_type(16))) _Float16 v16h;
typedef __attribute__((ext_vector_type(8)))  _Float16 h8;
typedef __attribute__((ext_vector_type(4)))  _Float16 h4;
typedef __attribute__((ext_vector_type(8)))  float    v8f;

#define ATTN_B   8
#define ATTN_S   4096
#define ATTN_T   512
#define ATTN_D   1024
#define TTILE    16
#define NTHREADS 256          // 8 wave32 waves
#define NWAVES   (NTHREADS / 32)
#define SB1      128          // S-block for score phase (16 cols per wave)
#define SBC      64           // S-block for P@V phase
#define QPITCH   (ATTN_D + 8) // halves; pad => lanes hit distinct banks
#define CPITCH   (SBC + 8)    // 72 halves = 144B rows, 16B aligned
#define PFPITCH  (ATTN_S + 8) // persistent f16 p tile pitch (8208B rows, 16B aligned)
#define SMALL_BYTES 2048
#define SMEM_BYTES  (SMALL_BYTES + (TTILE + SB1) * QPITCH * 2)   // 299,264 B (phase-1 peak)

union H16 { v16h v; h8 h[2]; };
static __device__ inline v16h pack16(h8 lo, h8 hi) { H16 u; u.h[0] = lo; u.h[1] = hi; return u.v; }

static __device__ inline v8f wmma_f16(v16h a, v16h b, v8f c) {
  // D = A(16x32 f16) * B(32x16 f16) + C(16x16 f32)
  return __builtin_amdgcn_wmma_f32_16x16x32_f16(false, a, false, b, (short)0, c, false, false);
}

__global__ __launch_bounds__(NTHREADS)
void DotProductAttention_kernel(const float* __restrict__ ctx,
                                const float* __restrict__ query,
                                const int*   __restrict__ mask,
                                float* __restrict__ eout,   // [B,T,D]
                                float* __restrict__ pbuf)   // [B,T,S] (scratch then final p)
{
  extern __shared__ char smem[];
  float* red    = (float*)smem;            // 256 floats: cross-lane reductions
  float* rowmax = red + NTHREADS;          // 16
  float* rowsum = rowmax + TTILE;          // 16
  // phase 0/1 layout
  _Float16* qt  = (_Float16*)(smem + SMALL_BYTES);       // [16][QPITCH] f16 Q tile
  _Float16* kt  = qt + TTILE * QPITCH;                   // [SB1][QPITCH] f16 ctx block
  // phase 2/3 layout (reuses the same region; qt/kt are dead by then)
  _Float16* pf16 = (_Float16*)(smem + SMALL_BYTES);      // [16][PFPITCH] persistent f16 p tile
  _Float16* ktt  = pf16 + TTILE * PFPITCH;               // [D][CPITCH] ctx^T block

  const int tid  = threadIdx.x;
  const int wave = tid >> 5;
  const int lane = tid & 31;
  const int m    = lane & 15;          // A row / B column owned by this lane
  const int off  = (lane >> 4) * 8;    // half-wave K offset per 16-bit WMMA layout
  const int b    = blockIdx.x / (ATTN_T / TTILE);
  const int t0   = (blockIdx.x % (ATTN_T / TTILE)) * TTILE;

  const size_t ctx_base = (size_t)b * ATTN_S * ATTN_D;
  const size_t q_base   = ((size_t)b * ATTN_T + t0) * ATTN_D;
  const size_t p_base   = ((size_t)b * ATTN_T + t0) * ATTN_S;

  // ---------------- Phase 0: Q tile f32 -> f16 LDS ----------------
  for (int i = tid; i < TTILE * (ATTN_D / 4); i += NTHREADS) {
    int r  = i >> 8;            // D/4 = 256 float4 per row
    int c4 = i & 255;
    const float4 v = *(const float4*)(query + q_base + (size_t)r * ATTN_D + c4 * 4);
    h4 hv; hv.x = (_Float16)v.x; hv.y = (_Float16)v.y; hv.z = (_Float16)v.z; hv.w = (_Float16)v.w;
    *(h4*)(qt + r * QPITCH + c4 * 4) = hv;
  }
  __syncthreads();

  // ---------------- Phase 1: scores = (Q K^T)/sqrt(D), masked, -> pbuf ----------------
  for (int s0 = 0; s0 < ATTN_S; s0 += SB1) {
    for (int i = tid; i < SB1 * (ATTN_D / 4); i += NTHREADS) {
      int r  = i >> 8;
      int c4 = i & 255;
      const float* gp = ctx + ctx_base + (size_t)(s0 + r) * ATTN_D + c4 * 4;
      const float4 v = *(const float4*)gp;
      if (s0 + SB1 < ATTN_S) __builtin_prefetch(gp + (size_t)SB1 * ATTN_D); // global_prefetch_b8
      h4 hv; hv.x = (_Float16)v.x; hv.y = (_Float16)v.y; hv.z = (_Float16)v.z; hv.w = (_Float16)v.w;
      *(h4*)(kt + r * QPITCH + c4 * 4) = hv;
    }
    __syncthreads();

    const int scol = s0 + wave * TTILE;          // this wave's 16 source columns
    const _Float16* arow = qt + m * QPITCH;      // A: Q rows
    const _Float16* brow = kt + (wave * TTILE + m) * QPITCH; // B: ctx row = column n of B
    v8f acc = {};
    #pragma unroll 4
    for (int k = 0; k < ATTN_D; k += 32) {
      h8 a0 = *(const h8*)(arow + k + off);
      h8 a1 = *(const h8*)(arow + k + 16 + off);
      h8 b0 = *(const h8*)(brow + k + off);
      h8 b1 = *(const h8*)(brow + k + 16 + off);
      acc = wmma_f16(pack16(a0, a1), pack16(b0, b1), acc);
    }
    // epilogue: scale + mask, store raw scores (C layout: lane -> col N=m, VGPR r -> row r+8*(lane>>4))
    const int sg = scol + m;
    const int mk = mask[b * ATTN_S + sg];
    const int mrow = (lane >> 4) * 8;
    #pragma unroll
    for (int r = 0; r < 8; r++) {
      float v = mk ? acc[r] * 0.03125f : -10000.0f;
      pbuf[p_base + (size_t)(mrow + r) * ATTN_S + sg] = v;
    }
    __syncthreads();
  }
  __threadfence();
  __syncthreads();

  // ---------------- Phase 2: softmax over S (float4 passes on L2-hot block) ----------------
  const int rr = tid >> 4;  // row 0..15
  const int cc = tid & 15;
  float* prow = pbuf + p_base + (size_t)rr * ATTN_S;
  const float4* prow4 = (const float4*)prow;

  float mx = -3.4e38f;
  for (int j = cc; j < ATTN_S / 4; j += 16) {
    float4 v = prow4[j];
    mx = fmaxf(mx, fmaxf(fmaxf(v.x, v.y), fmaxf(v.z, v.w)));
  }
  red[tid] = mx;
  __syncthreads();
  if (tid < TTILE) {
    float v = red[tid * 16];
    #pragma unroll
    for (int j = 1; j < 16; j++) v = fmaxf(v, red[tid * 16 + j]);
    rowmax[tid] = v;
  }
  __syncthreads();

  const float rmax = rowmax[rr];
  float sm = 0.0f;
  for (int j = cc; j < ATTN_S / 4; j += 16) {
    float4 v = prow4[j];
    sm += expf(v.x - rmax) + expf(v.y - rmax) + expf(v.z - rmax) + expf(v.w - rmax);
  }
  red[tid] = sm;
  __syncthreads();
  if (tid < TTILE) {
    float v = 0.0f;
    #pragma unroll
    for (int j = 0; j < 16; j++) v += red[tid * 16 + j];
    rowsum[tid] = v;
  }
  __syncthreads();

  // normalize: write final f32 p to global AND persistent f16 p tile to LDS
  const float inv = 1.0f / rowsum[rr];
  _Float16* pfrow = pf16 + rr * PFPITCH;
  for (int j = cc; j < ATTN_S / 4; j += 16) {
    float4 v = prow4[j];
    float4 o;
    o.x = expf(v.x - rmax) * inv; o.y = expf(v.y - rmax) * inv;
    o.z = expf(v.z - rmax) * inv; o.w = expf(v.w - rmax) * inv;
    *(float4*)(prow + j * 4) = o;
    h4 hv; hv.x = (_Float16)o.x; hv.y = (_Float16)o.y; hv.z = (_Float16)o.z; hv.w = (_Float16)o.w;
    *(h4*)(pfrow + j * 4) = hv;
  }
  __syncthreads();

  // ---------------- Phase 3: expected_ctx = P @ ctx ----------------
  v8f acc[8];
  #pragma unroll
  for (int j = 0; j < 8; j++) acc[j] = (v8f){};
  const int dbase = wave * 128;   // 8 d-subtiles of 16 per wave
  const _Float16* parow = pf16 + m * PFPITCH;   // this lane's A row (persistent in LDS)

  for (int s0 = 0; s0 < ATTN_S; s0 += SBC) {
    // stage ctx block transposed: ktt[d][s_local] f16 (coalesced global float4 reads)
    for (int i = tid; i < SBC * (ATTN_D / 4); i += NTHREADS) {
      int dg = i & 255;           // float4 index along D
      int sl = i >> 8;            // 0..63
      const float4 v = *(const float4*)(ctx + ctx_base + (size_t)(s0 + sl) * ATTN_D + dg * 4);
      ktt[(dg * 4 + 0) * CPITCH + sl] = (_Float16)v.x;
      ktt[(dg * 4 + 1) * CPITCH + sl] = (_Float16)v.y;
      ktt[(dg * 4 + 2) * CPITCH + sl] = (_Float16)v.z;
      ktt[(dg * 4 + 3) * CPITCH + sl] = (_Float16)v.w;
    }
    __syncthreads();

    #pragma unroll
    for (int ks = 0; ks < SBC; ks += 32) {
      h8 a0 = *(const h8*)(parow + s0 + ks + off);
      h8 a1 = *(const h8*)(parow + s0 + ks + 16 + off);
      v16h a = pack16(a0, a1);
      #pragma unroll
      for (int j = 0; j < 8; j++) {
        const _Float16* bp = ktt + (dbase + j * 16 + m) * CPITCH + ks;
        h8 b0 = *(const h8*)(bp + off);
        h8 b1 = *(const h8*)(bp + 16 + off);
        acc[j] = wmma_f16(a, pack16(b0, b1), acc[j]);
      }
    }
    __syncthreads();
  }

  // write expected_ctx (coalesced over N=lane&15 within half-waves)
  const int mrow = (lane >> 4) * 8;
  #pragma unroll
  for (int j = 0; j < 8; j++)
    #pragma unroll
    for (int r = 0; r < 8; r++)
      eout[((size_t)b * ATTN_T + t0 + mrow + r) * ATTN_D + dbase + j * 16 + m] = acc[j][r];
}

extern "C" void kernel_launch(void* const* d_in, const int* in_sizes, int n_in,
                              void* d_out, int out_size, void* d_ws, size_t ws_size,
                              hipStream_t stream) {
  (void)in_sizes; (void)n_in; (void)out_size; (void)d_ws; (void)ws_size;
  const float* ctx   = (const float*)d_in[0];
  const float* query = (const float*)d_in[1];
  const int*   mask  = (const int*)d_in[2];
  float* eout = (float*)d_out;                                   // [B,T,D] first
  float* pbuf = eout + (size_t)ATTN_B * ATTN_T * ATTN_D;         // then [B,T,S]
  dim3 grid(ATTN_B * (ATTN_T / TTILE));   // 256 workgroups
  dim3 block(NTHREADS);                   // 8 waves
  DotProductAttention_kernel<<<grid, block, SMEM_BYTES, stream>>>(ctx, query, mask, eout, pbuf);
}
